// MoeLayer_6073083756562
// MI455X (gfx1250) — matrix operand order repaired and checked by reading, not compiled, not run on hardware
//
#include <hip/hip_runtime.h>
#include <stdint.h>

// Problem constants (from reference)
#define T_TOK 2048
#define D_DIM 1024
#define F_DIM 3584
#define E_NUM 8
// TOPK = 2 hardcoded throughout

typedef __attribute__((ext_vector_type(16))) __bf16 v16bf;
typedef __attribute__((ext_vector_type(8)))  float  v8f;

union Frag {
    v16bf v;
    uint4 q[2];
};

__device__ __forceinline__ unsigned short f2bf_bits(float f) {
    union { float f; unsigned u; } v;
    v.f = f;
    unsigned r = (v.u + 0x7FFFu + ((v.u >> 16) & 1u)) >> 16;  // RNE
    return (unsigned short)r;
}

// round two floats to bf16 (RNE) and pack into one dword: {hi.bf16, lo.bf16}
__device__ __forceinline__ unsigned pack2bf(float lo, float hi) {
    union { float f; unsigned u; } a, b;
    a.f = lo; b.f = hi;
    unsigned ar = a.u + 0x7FFFu + ((a.u >> 16) & 1u);
    unsigned br = b.u + 0x7FFFu + ((b.u >> 16) & 1u);
    return __builtin_amdgcn_perm(br, ar, 0x07060302u);  // {br[3],br[2],ar[3],ar[2]}
}

// ---------------------------------------------------------------- init
__global__ void moe_init(int* cnt) {
    if (threadIdx.x < E_NUM) cnt[threadIdx.x] = 0;
}

// ---------------------------------------------------------------- gating
__global__ __launch_bounds__(256)
void moe_gate(const float* __restrict__ x, const float* __restrict__ gw,
              int* __restrict__ tok_e, float* __restrict__ tok_w,
              int* __restrict__ cnt) {
    const int wave = threadIdx.x >> 5;
    const int lane = threadIdx.x & 31;
    const int t = blockIdx.x * 8 + wave;

    float acc[E_NUM];
#pragma unroll
    for (int e = 0; e < E_NUM; ++e) acc[e] = 0.0f;

    for (int d = lane; d < D_DIM; d += 32) {
        float xv = x[(size_t)t * D_DIM + d];
        const float* g = gw + (size_t)d * E_NUM;
#pragma unroll
        for (int e = 0; e < E_NUM; ++e) acc[e] += xv * g[e];
    }
#pragma unroll
    for (int e = 0; e < E_NUM; ++e) {
#pragma unroll
        for (int o = 16; o >= 1; o >>= 1)
            acc[e] += __shfl_xor(acc[e], o, 32);
    }
    if (lane == 0) {
        float m1 = acc[0], m2 = -3.0e38f;
        int i1 = 0, i2 = 0;
#pragma unroll
        for (int e = 1; e < E_NUM; ++e) {
            float v = acc[e];
            if (v > m1) { m2 = m1; i2 = i1; m1 = v; i1 = e; }
            else if (v > m2) { m2 = v; i2 = e; }
        }
        float b = __expf(m2 - m1);
        float s = 1.0f + b;
        tok_e[2 * t + 0] = i1;
        tok_e[2 * t + 1] = i2;
        tok_w[2 * t + 0] = 1.0f / s;
        tok_w[2 * t + 1] = b / s;
        atomicAdd(&cnt[i1], 1);
        atomicAdd(&cnt[i2], 1);
    }
}

// ---------------------------------------------------------------- offsets
__global__ void moe_offsets(const int* __restrict__ cnt, int* __restrict__ off,
                            int* __restrict__ cur) {
    if (threadIdx.x == 0) {
        int s = 0;
        for (int e = 0; e < E_NUM; ++e) { off[e] = s; cur[e] = s; s += cnt[e]; }
    }
}

// ---------------------------------------------------------------- scatter
__global__ __launch_bounds__(256)
void moe_scatter(const int* __restrict__ tok_e, int* __restrict__ cur,
                 const int* __restrict__ off, int* __restrict__ rows_tok,
                 int* __restrict__ tok_row) {
    int t = blockIdx.x * 256 + threadIdx.x;
    if (t >= T_TOK) return;
#pragma unroll
    for (int k = 0; k < 2; ++k) {
        int e = tok_e[2 * t + k];
        int p = atomicAdd(&cur[e], 1);
        int r = off[e] + p;
        rows_tok[r] = t;
        tok_row[2 * t + k] = r;
    }
}

// ---------------------------------------------------------------- GEMM1
// per (F-tile 64, token-tile 128, expert): H = silu(X W1) * (X W3), bf16 out
#define LDSR 56  // LDS row stride in halfs (112 B: 16B aligned, conflict-free)

__global__ __launch_bounds__(256)
void moe_gemm1(const float* __restrict__ x, const float* __restrict__ w1,
               const float* __restrict__ w3, const int* __restrict__ cnt,
               const int* __restrict__ off, const int* __restrict__ rows_tok,
               unsigned short* __restrict__ H) {
    const int e = blockIdx.z;
    const int ne = cnt[e];
    const int rowBase = blockIdx.y * 128;
    if (rowBase >= ne) return;
    const int base = off[e];
    const int fBase = blockIdx.x * 64;

    __shared__ unsigned short smA[2][128 * LDSR];
    __shared__ unsigned short smB1[2][64 * LDSR];
    __shared__ unsigned short smB3[2][64 * LDSR];
    __shared__ int smTok[128];

    const int tid = threadIdx.x;
    if (tid < 128) {
        int r = rowBase + tid;
        int rc = (r < ne) ? r : (ne - 1);
        smTok[tid] = rows_tok[base + rc];
    }
    __syncthreads();

    const int lane = tid & 31;
    const int wave = tid >> 5;
    const int wRow = (wave & 3) * 32;    // 4 row groups of 32 (2 m-tiles each)
    const int wCol = (wave >> 2) * 32;   // 2 col groups of 32 (2 n-tiles each)
    const int half = (lane >= 16) ? 1 : 0;
    const int l16 = lane & 15;

    // loader indices
    const int aRow = tid >> 1;           // 0..127
    const int aK   = (tid & 1) * 16;     // 0 or 16
    const int bK2  = (tid & 15) * 2;     // even k: 0..30
    const int bN   = (tid >> 4) * 4;     // 0..60

    const float* w1p = w1 + (size_t)e * D_DIM * F_DIM;
    const float* w3p = w3 + (size_t)e * D_DIM * F_DIM;
    const float* xrow = x + (size_t)smTok[aRow] * D_DIM + aK;

    v8f acc1[2][2] = {};
    v8f acc3[2][2] = {};

    float4 ra0, ra1, ra2, ra3, rb10, rb11, rb30, rb31;

    auto fetch = [&](int kk) {
        const float4* sa = (const float4*)(xrow + kk);
        ra0 = sa[0]; ra1 = sa[1]; ra2 = sa[2]; ra3 = sa[3];
        size_t g0 = (size_t)(kk + bK2) * F_DIM + fBase + bN;
        rb10 = *(const float4*)(w1p + g0);
        rb11 = *(const float4*)(w1p + g0 + F_DIM);
        rb30 = *(const float4*)(w3p + g0);
        rb31 = *(const float4*)(w3p + g0 + F_DIM);
    };
    auto stash = [&](unsigned short* A, unsigned short* B1, unsigned short* B3) {
        unsigned* da = (unsigned*)(A + aRow * LDSR + aK);
        da[0] = pack2bf(ra0.x, ra0.y); da[1] = pack2bf(ra0.z, ra0.w);
        da[2] = pack2bf(ra1.x, ra1.y); da[3] = pack2bf(ra1.z, ra1.w);
        da[4] = pack2bf(ra2.x, ra2.y); da[5] = pack2bf(ra2.z, ra2.w);
        da[6] = pack2bf(ra3.x, ra3.y); da[7] = pack2bf(ra3.z, ra3.w);
        *(unsigned*)(B1 + (bN + 0) * LDSR + bK2) = pack2bf(rb10.x, rb11.x);
        *(unsigned*)(B1 + (bN + 1) * LDSR + bK2) = pack2bf(rb10.y, rb11.y);
        *(unsigned*)(B1 + (bN + 2) * LDSR + bK2) = pack2bf(rb10.z, rb11.z);
        *(unsigned*)(B1 + (bN + 3) * LDSR + bK2) = pack2bf(rb10.w, rb11.w);
        *(unsigned*)(B3 + (bN + 0) * LDSR + bK2) = pack2bf(rb30.x, rb31.x);
        *(unsigned*)(B3 + (bN + 1) * LDSR + bK2) = pack2bf(rb30.y, rb31.y);
        *(unsigned*)(B3 + (bN + 2) * LDSR + bK2) = pack2bf(rb30.z, rb31.z);
        *(unsigned*)(B3 + (bN + 3) * LDSR + bK2) = pack2bf(rb30.w, rb31.w);
    };
    auto mma = [&](const unsigned short* A, const unsigned short* B1,
                   const unsigned short* B3) {
        Frag a[2], b1[2], b3[2];
#pragma unroll
        for (int mt = 0; mt < 2; ++mt) {
            // 16-bit A 16x32: lanes<16 hold K 0-7 & 16-23, lanes>=16 hold 8-15 & 24-31
            const unsigned short* pa = A + (wRow + mt * 16 + l16) * LDSR + half * 8;
            a[mt].q[0] = *(const uint4*)(pa);
            a[mt].q[1] = *(const uint4*)(pa + 16);
        }
#pragma unroll
        for (int nt = 0; nt < 2; ++nt) {
            // 16-bit B 32x16: lane holds col n=lane%16; lanes<16 K 0-7 & 16-23, lanes>=16 K 8-15 & 24-31
            int n = wCol + nt * 16 + l16;
            const unsigned short* p1 = B1 + n * LDSR + half * 8;
            const unsigned short* p3 = B3 + n * LDSR + half * 8;
            b1[nt].q[0] = *(const uint4*)(p1);
            b1[nt].q[1] = *(const uint4*)(p1 + 16);
            b3[nt].q[0] = *(const uint4*)(p3);
            b3[nt].q[1] = *(const uint4*)(p3 + 16);
        }
#pragma unroll
        for (int mt = 0; mt < 2; ++mt) {
#pragma unroll
            for (int nt = 0; nt < 2; ++nt) {
                acc1[mt][nt] = __builtin_amdgcn_wmma_f32_16x16x32_bf16(
                    false, a[mt].v, false, b1[nt].v, (short)0, acc1[mt][nt], false, false);
                acc3[mt][nt] = __builtin_amdgcn_wmma_f32_16x16x32_bf16(
                    false, a[mt].v, false, b3[nt].v, (short)0, acc3[mt][nt], false, false);
            }
        }
    };

    // software pipeline, 2x unrolled so all LDS bases are compile-time constant
    fetch(0);
    stash(smA[0], smB1[0], smB3[0]);
    __syncthreads();

    const int NK = D_DIM / 32;  // 32, even
#pragma unroll 1
    for (int i = 0; i < NK; i += 2) {
        fetch((i + 1) * 32);
        mma(smA[0], smB1[0], smB3[0]);
        stash(smA[1], smB1[1], smB3[1]);
        __syncthreads();

        const bool more = (i + 2) < NK;
        if (more) fetch((i + 2) * 32);
        mma(smA[1], smB1[1], smB3[1]);
        if (more) stash(smA[0], smB1[0], smB3[0]);
        __syncthreads();
    }

    // SwiGLU + store H (bf16). C/D layout: VGPR v -> row v (lanes<16) / 8+v (lanes>=16)
#pragma unroll
    for (int mt = 0; mt < 2; ++mt) {
#pragma unroll
        for (int nt = 0; nt < 2; ++nt) {
#pragma unroll
            for (int v = 0; v < 8; ++v) {
                int m = wRow + mt * 16 + v + half * 8;
                int r = rowBase + m;
                if (r < ne) {
                    float h1 = acc1[mt][nt][v];
                    float h3 = acc3[mt][nt][v];
                    float sig = __builtin_amdgcn_rcpf(1.0f + __expf(-h1));
                    int col = fBase + wCol + nt * 16 + l16;
                    H[(size_t)(base + r) * F_DIM + col] = f2bf_bits(h1 * sig * h3);
                }
            }
        }
    }
}

// ---------------------------------------------------------------- GEMM2
// per (D-tile 64, token-tile 128, expert): stage = H x W2 (fp32 out)
__global__ __launch_bounds__(256)
void moe_gemm2(const unsigned short* __restrict__ H, const float* __restrict__ w2,
               const int* __restrict__ cnt, const int* __restrict__ off,
               float* __restrict__ stage) {
    const int e = blockIdx.z;
    const int ne = cnt[e];
    const int rowBase = blockIdx.y * 128;
    if (rowBase >= ne) return;
    const int base = off[e];
    const int dBase = blockIdx.x * 64;

    __shared__ unsigned short smA[2][128 * LDSR];
    __shared__ unsigned short smB[2][64 * LDSR];

    const int tid = threadIdx.x;
    const int lane = tid & 31;
    const int wave = tid >> 5;
    const int wRow = (wave & 3) * 32;
    const int wCol = (wave >> 2) * 32;
    const int half = (lane >= 16) ? 1 : 0;
    const int l16 = lane & 15;

    const int aRow = tid >> 1;           // 0..127
    const int aK   = (tid & 1) * 16;     // 0 or 16
    const int bK2  = (tid & 15) * 2;
    const int bN   = (tid >> 4) * 4;

    int arc = rowBase + aRow;
    arc = (arc < ne) ? arc : (ne - 1);
    const unsigned short* hrow = H + (size_t)(base + arc) * F_DIM + aK;
    const float* w2p = w2 + (size_t)e * F_DIM * D_DIM;

    v8f acc[2][2] = {};

    uint4 ra0, ra1;
    float4 rb0, rb1;

    auto fetch = [&](int kk) {
        ra0 = *(const uint4*)(hrow + kk);
        ra1 = *(const uint4*)(hrow + kk + 8);
        size_t g0 = (size_t)(kk + bK2) * D_DIM + dBase + bN;
        rb0 = *(const float4*)(w2p + g0);
        rb1 = *(const float4*)(w2p + g0 + D_DIM);
    };
    auto stash = [&](unsigned short* A, unsigned short* B) {
        *(uint4*)(A + aRow * LDSR + aK) = ra0;
        *(uint4*)(A + aRow * LDSR + aK + 8) = ra1;
        *(unsigned*)(B + (bN + 0) * LDSR + bK2) = pack2bf(rb0.x, rb1.x);
        *(unsigned*)(B + (bN + 1) * LDSR + bK2) = pack2bf(rb0.y, rb1.y);
        *(unsigned*)(B + (bN + 2) * LDSR + bK2) = pack2bf(rb0.z, rb1.z);
        *(unsigned*)(B + (bN + 3) * LDSR + bK2) = pack2bf(rb0.w, rb1.w);
    };
    auto mma = [&](const unsigned short* A, const unsigned short* B) {
        Frag a[2], b[2];
#pragma unroll
        for (int mt = 0; mt < 2; ++mt) {
            const unsigned short* pa = A + (wRow + mt * 16 + l16) * LDSR + half * 8;
            a[mt].q[0] = *(const uint4*)(pa);
            a[mt].q[1] = *(const uint4*)(pa + 16);
        }
#pragma unroll
        for (int nt = 0; nt < 2; ++nt) {
            int n = wCol + nt * 16 + l16;
            const unsigned short* pb = B + n * LDSR + half * 8;
            b[nt].q[0] = *(const uint4*)(pb);
            b[nt].q[1] = *(const uint4*)(pb + 16);
        }
#pragma unroll
        for (int mt = 0; mt < 2; ++mt) {
#pragma unroll
            for (int nt = 0; nt < 2; ++nt) {
                acc[mt][nt] = __builtin_amdgcn_wmma_f32_16x16x32_bf16(
                    false, a[mt].v, false, b[nt].v, (short)0, acc[mt][nt], false, false);
            }
        }
    };

    fetch(0);
    stash(smA[0], smB[0]);
    __syncthreads();

    const int NK = F_DIM / 32;  // 112, even
#pragma unroll 1
    for (int i = 0; i < NK; i += 2) {
        fetch((i + 1) * 32);
        mma(smA[0], smB[0]);
        stash(smA[1], smB[1]);
        __syncthreads();

        const bool more = (i + 2) < NK;
        if (more) fetch((i + 2) * 32);
        mma(smA[1], smB[1]);
        if (more) stash(smA[0], smB[0]);
        __syncthreads();
    }

#pragma unroll
    for (int mt = 0; mt < 2; ++mt) {
#pragma unroll
        for (int nt = 0; nt < 2; ++nt) {
#pragma unroll
            for (int v = 0; v < 8; ++v) {
                int m = wRow + mt * 16 + v + half * 8;
                int r = rowBase + m;
                if (r < ne) {
                    int col = dBase + wCol + nt * 16 + l16;
                    stage[(size_t)(base + r) * D_DIM + col] = acc[mt][nt][v];
                }
            }
        }
    }
}

// ---------------------------------------------------------------- combine
__global__ __launch_bounds__(256)
void moe_combine(const float* __restrict__ stage, const int* __restrict__ tok_row,
                 const float* __restrict__ tok_w, float* __restrict__ out) {
    int idx = blockIdx.x * 256 + threadIdx.x;   // over T*D/4
    int t  = idx >> 8;                          // D/4 == 256
    int dq = idx & 255;
    int r0 = tok_row[2 * t], r1 = tok_row[2 * t + 1];
    float w0 = tok_w[2 * t], w1 = tok_w[2 * t + 1];
    float4 s0 = ((const float4*)(stage + (size_t)r0 * D_DIM))[dq];
    float4 s1 = ((const float4*)(stage + (size_t)r1 * D_DIM))[dq];
    float4 o;
    o.x = w0 * s0.x + w1 * s1.x;
    o.y = w0 * s0.y + w1 * s1.y;
    o.z = w0 * s0.z + w1 * s1.z;
    o.w = w0 * s0.w + w1 * s1.w;
    ((float4*)(out + (size_t)t * D_DIM))[dq] = o;
}

// ---------------------------------------------------------------- launch
extern "C" void kernel_launch(void* const* d_in, const int* in_sizes, int n_in,
                              void* d_out, int out_size, void* d_ws, size_t ws_size,
                              hipStream_t stream) {
    const float* x  = (const float*)d_in[0];
    const float* gw = (const float*)d_in[1];
    const float* w1 = (const float*)d_in[2];
    const float* w3 = (const float*)d_in[3];
    const float* w2 = (const float*)d_in[4];
    float* out = (float*)d_out;

    uintptr_t p = (uintptr_t)d_ws;
    auto alloc = [&](size_t bytes) -> void* {
        void* r = (void*)p;
        p += (bytes + 255) & ~(size_t)255;
        return r;
    };
    int* cnt      = (int*)alloc(E_NUM * sizeof(int));
    int* off      = (int*)alloc(E_NUM * sizeof(int));
    int* cur      = (int*)alloc(E_NUM * sizeof(int));
    int* tok_e    = (int*)alloc(2 * T_TOK * sizeof(int));
    float* tok_w  = (float*)alloc(2 * T_TOK * sizeof(float));
    int* rows_tok = (int*)alloc((2 * T_TOK + 128) * sizeof(int));
    int* tok_row  = (int*)alloc(2 * T_TOK * sizeof(int));
    unsigned short* H = (unsigned short*)alloc((size_t)(2 * T_TOK) * F_DIM * sizeof(unsigned short));
    float* stage  = (float*)alloc((size_t)(2 * T_TOK) * D_DIM * sizeof(float));

    moe_init<<<1, 32, 0, stream>>>(cnt);
    moe_gate<<<T_TOK / 8, 256, 0, stream>>>(x, gw, tok_e, tok_w, cnt);
    moe_offsets<<<1, 32, 0, stream>>>(cnt, off, cur);
    moe_scatter<<<T_TOK / 256, 256, 0, stream>>>(tok_e, cur, off, rows_tok, tok_row);
    moe_gemm1<<<dim3(F_DIM / 64, T_TOK / 128, E_NUM), 256, 0, stream>>>(
        x, w1, w3, cnt, off, rows_tok, H);
    moe_gemm2<<<dim3(D_DIM / 64, T_TOK / 128, E_NUM), 256, 0, stream>>>(
        H, w2, cnt, off, stage);
    moe_combine<<<(T_TOK * D_DIM / 4) / 256, 256, 0, stream>>>(
        stage, tok_row, tok_w, out);
}
